// WindowAttention_4715874091442
// MI455X (gfx1250) — compile-verified, matrix operand order checked by reading, not run on hardware
//
#include <hip/hip_runtime.h>
#include <hip/hip_bf16.h>

// ---------------------------------------------------------------------------
// Swin shifted-window attention for MI455X (gfx1250), bf16 WMMA pipeline.
// Fragment-major LDS layouts: every WMMA operand is loaded with two
// ds_load_b128 ops (32 contiguous bytes per lane), no per-element repacking.
// ---------------------------------------------------------------------------

typedef __bf16 bf16;
typedef __attribute__((ext_vector_type(16))) __bf16 v16bf;
typedef __attribute__((ext_vector_type(8)))  float  v8f;

#define BDIM   8
#define NTOK   4096
#define DMODEL 512
#define NHEAD  8
#define DHEAD  64
#define WWIN   128
#define PWIN   32     // number of windows = NTOK / WWIN
#define DISP   64     // cyclic shift

// float -> bf16 bits, round-to-nearest-even
__device__ __forceinline__ unsigned short bfbits(float f) {
  union { float f; unsigned u; } v; v.f = f;
  return (unsigned short)((v.u + 0x7fffu + ((v.u >> 16) & 1u)) >> 16);
}

// Inverse fragment mapping for 16-bit WMMA operands (ISA 7.12.2):
// element with K index kk (0..31) and row/col index mn (0..15) lives in
//   lane = mn + 16*((kk>>3)&1),  half j = ((kk>>4)<<3) | (kk&7)
__device__ __forceinline__ int frag_j(int kk)  { return ((kk >> 4) << 3) | (kk & 7); }
__device__ __forceinline__ int frag_lh(int kk) { return ((kk >> 3) & 1) << 4; }

__device__ __forceinline__ v8f wmma_bf16(v16bf a, v16bf b, v8f c) {
  return __builtin_amdgcn_wmma_f32_16x16x32_bf16(
      /*neg_a=*/false, a, /*neg_b=*/false, b,
      /*c_mod=*/(short)0, c, /*reuse_a=*/false, /*reuse_b=*/false);
}

__device__ __forceinline__ v16bf ld_frag(const unsigned short* p) {
  return *(const v16bf*)p;   // 32B aligned -> 2x ds_load_b128
}

// ---------------------------------------------------------------------------
// Kernel 0: f32 -> bf16 weight conversion (one-time, tiny)
// ---------------------------------------------------------------------------
__global__ __launch_bounds__(256) void swin_cvt_bf16(
    const float* __restrict__ s, unsigned short* __restrict__ d, int n) {
  int i = blockIdx.x * 256 + threadIdx.x;
  if (i < n) d[i] = bfbits(s[i]);
}

// ---------------------------------------------------------------------------
// Kernel 1: qkv = roll(x,-DISP) @ Wqkv   [32768 x 512] * [512 x 1536]
// grid = (12 col-tiles, 256 row-tiles), block = 256 (8 waves).
// Each wave: 16 rows x 128 cols = 8 accumulator tiles, K stepped by 32.
// A staged fragment-major per (wave,lane); B fragment-major per (tile,lane).
// ---------------------------------------------------------------------------
__global__ __launch_bounds__(256) void swin_qkv_gemm(
    const float* __restrict__ x, const unsigned short* __restrict__ wq,
    unsigned short* __restrict__ qw, unsigned short* __restrict__ kw,
    unsigned short* __restrict__ vw)
{
  __shared__ __align__(32) unsigned short As[8 * 32 * 16];  // 8KB, frag-major
  __shared__ __align__(32) unsigned short Bs[8 * 32 * 16];  // 8KB, frag-major
  const int tid = threadIdx.x, lane = tid & 31, wv = tid >> 5;
  const int ntile = blockIdx.x;          // 0..11 (cols of 1536)
  const int mtile = blockIdx.y;          // 0..255 (rows of 32768)
  const int b  = (mtile * 128) / NTOK;
  const int n0 = (mtile * 128) % NTOK;

  v8f acc[8];
  for (int t = 0; t < 8; ++t)
    for (int e = 0; e < 8; ++e) acc[t][e] = 0.0f;

  for (int k0 = 0; k0 < DMODEL; k0 += 32) {
    // Stage A: 128 tokens x 32 K; float2 load -> packed b32 LDS store.
    for (int i = tid; i < 128 * 16; i += 256) {
      int r = i >> 4, c = (i & 15) * 2;
      int src = (n0 + r + DISP) & (NTOK - 1);
      float2 f = *(const float2*)&x[((size_t)b * NTOK + src) * DMODEL + k0 + c];
      unsigned pk = (unsigned)bfbits(f.x) | ((unsigned)bfbits(f.y) << 16);
      int dst = (((r >> 4) * 32) + (r & 15) + frag_lh(c)) * 16 + frag_j(c);
      *(unsigned*)&As[dst] = pk;
    }
    // Stage B: 32 K x 128 cols; b64 global load (4 bf16) -> 4 b16 LDS stores.
    for (int i = tid; i < 32 * 32; i += 256) {
      int c = i >> 5, jq = (i & 31) * 4;
      unsigned long long w4 =
          *(const unsigned long long*)&wq[(size_t)(k0 + c) * 1536 + ntile * 128 + jq];
      int j = frag_j(c), lh = frag_lh(c);
#pragma unroll
      for (int e = 0; e < 4; ++e) {
        int jc = jq + e;
        Bs[(((jc >> 4) * 32) + (jc & 15) + lh) * 16 + j] =
            (unsigned short)(w4 >> (16 * e));
      }
    }
    if (k0 + 32 < DMODEL) {   // prefetch next A K-slice
      int r = tid >> 1;
      int src = (n0 + r + DISP) & (NTOK - 1);
      __builtin_prefetch(
          &x[((size_t)b * NTOK + src) * DMODEL + k0 + 32 + (tid & 1) * 16], 0, 0);
    }
    __syncthreads();

    const v16bf af = ld_frag(&As[(wv * 32 + lane) * 16]);
#pragma unroll
    for (int t = 0; t < 8; ++t) {
      const v16bf bfr = ld_frag(&Bs[(t * 32 + lane) * 16]);
      acc[t] = wmma_bf16(af, bfr, acc[t]);
    }
    __syncthreads();
  }

  // D layout: reg r, lane L -> row m = r + 8*(L>>4), col = (L&15).
  // Scatter into windowed layout [b][h][pp][w][dh] (bf16 bits).
  for (int t = 0; t < 8; ++t) {
    int col = ntile * 128 + t * 16 + (lane & 15);   // 0..1535
    int which = col >> 9;                           // 0=q 1=k 2=v
    int hd = col & 511;
    int h = hd >> 6, d = hd & 63;
    unsigned short* dst = (which == 0) ? qw : (which == 1) ? kw : vw;
    for (int r = 0; r < 8; ++r) {
      int m = r + ((lane >> 4) << 3);
      int n = n0 + wv * 16 + m;        // shifted token index
      int pp = n & 31, wn = n >> 5;    // window id, position in window
      dst[((((size_t)b * NHEAD + h) * PWIN + pp) * WWIN + wn) * DHEAD + d] =
          bfbits(acc[t][r]);
    }
  }
}

// ---------------------------------------------------------------------------
// Kernel 2: per-(b,h,window) attention.  2048 blocks x 256 threads (8 waves).
// LDS (frag-major): q 16K + k 16K + v 16K + pos 1K; probs overlay q+k.
// Each wave owns 16 query rows; softmax fully in registers via __shfl_xor
// across the 16-lane halves (matches C/D fragment layout).
// ---------------------------------------------------------------------------
__global__ __launch_bounds__(256) void swin_attn(
    const unsigned short* __restrict__ qw, const unsigned short* __restrict__ kw,
    const unsigned short* __restrict__ vw, const float* __restrict__ pos,
    unsigned short* __restrict__ ow)
{
  __shared__ __align__(32) unsigned char smem[50176];
  unsigned short* q_s = (unsigned short*)smem;            // [8wv][2ks][32][16]
  unsigned short* k_s = (unsigned short*)(smem + 16384);  // [8t][2ks][32][16]
  unsigned short* v_s = (unsigned short*)(smem + 32768);  // [4t][4ks][32][16]
  unsigned short* p_s = (unsigned short*)smem;            // [8wv][4ks][32][16]
  float* pos_s = (float*)(smem + 49152);                  // 255 entries

  const int blk = blockIdx.x;            // ((b*8 + h)*32 + pp)
  const int tid = threadIdx.x, lane = tid & 31, wv = tid >> 5;
  const size_t base = (size_t)blk * WWIN * DHEAD;

  // q: A-fragments (M=token, K=dh); k: B-fragments (K=dh, N=token).
  // Pairs along dh share dest lane and adjacent j -> b32 LDS stores.
  for (int i = tid; i < 128 * 32; i += 256) {
    int row = i >> 5, dh = (i & 31) * 2;
    int ks = dh >> 5, kk = dh & 31;
    int j = frag_j(kk), lh = frag_lh(kk);
    int lane_d = (row & 15) + lh;
    unsigned dq = *(const unsigned*)&qw[base + row * 64 + dh];
    *(unsigned*)&q_s[(((row >> 4) * 2 + ks) * 32 + lane_d) * 16 + j] = dq;
    unsigned dk = *(const unsigned*)&kw[base + row * 64 + dh];
    *(unsigned*)&k_s[(((row >> 4) * 2 + ks) * 32 + lane_d) * 16 + j] = dk;
  }
  // v: B-fragments (K=token, N=dh); b64 global load, 4 b16 scattered stores.
  for (int i = tid; i < 128 * 16; i += 256) {
    int row = i >> 4, d4 = (i & 15) * 4;
    unsigned long long w4 =
        *(const unsigned long long*)&vw[base + row * 64 + d4];
    int ks = row >> 5, kk = row & 31;
    int j = frag_j(kk), lh = frag_lh(kk);
#pragma unroll
    for (int e = 0; e < 4; ++e) {
      int dh = d4 + e;
      v_s[(((dh >> 4) * 4 + ks) * 32 + (dh & 15) + lh) * 16 + j] =
          (unsigned short)(w4 >> (16 * e));
    }
  }
  if (tid < 2 * WWIN - 1) pos_s[tid] = pos[tid];
  __syncthreads();

  // ---- S = q @ k^T (K = DHEAD = 64, two WMMA K-steps) ----
  v8f acc[8];
  for (int t = 0; t < 8; ++t)
    for (int e = 0; e < 8; ++e) acc[t][e] = 0.0f;

#pragma unroll
  for (int ks = 0; ks < 2; ++ks) {
    const v16bf af = ld_frag(&q_s[((wv * 2 + ks) * 32 + lane) * 16]);
#pragma unroll
    for (int t = 0; t < 8; ++t) {
      const v16bf bfr = ld_frag(&k_s[((t * 2 + ks) * 32 + lane) * 16]);
      acc[t] = wmma_bf16(af, bfr, acc[t]);
    }
  }

  // ---- scale + relative-position bias + shift mask, then softmax ----
  const float scale = 0.125f;   // DH^-0.5
  float rmax[8], rsum[8];
  for (int r = 0; r < 8; ++r) rmax[r] = -3.0e38f;
  for (int t = 0; t < 8; ++t)
    for (int r = 0; r < 8; ++r) {
      int i = wv * 16 + r + ((lane >> 4) << 3);   // query pos in window
      int j = t * 16 + (lane & 15);               // key pos in window
      float s = acc[t][r] * scale + pos_s[j - i + (WWIN - 1)];
      if ((i >= WWIN - DISP) != (j >= WWIN - DISP)) s -= 1.0e9f;
      acc[t][r] = s;
      rmax[r] = fmaxf(rmax[r], s);
    }
  for (int r = 0; r < 8; ++r)
    for (int off = 1; off < 16; off <<= 1)        // reduce within 16-lane half
      rmax[r] = fmaxf(rmax[r], __shfl_xor(rmax[r], off, 32));
  for (int r = 0; r < 8; ++r) rsum[r] = 0.0f;
  for (int t = 0; t < 8; ++t)
    for (int r = 0; r < 8; ++r) {
      float e = __expf(acc[t][r] - rmax[r]);
      acc[t][r] = e;
      rsum[r] += e;
    }
  for (int r = 0; r < 8; ++r) {
    for (int off = 1; off < 16; off <<= 1)
      rsum[r] += __shfl_xor(rsum[r], off, 32);
    rsum[r] = 1.0f / rsum[r];
  }

  __syncthreads();   // all waves finished reading q_s/k_s -> reuse as p_s

  // Write probs directly into A-fragment layout for this wave's 16 rows.
  for (int t = 0; t < 8; ++t)
    for (int r = 0; r < 8; ++r) {
      int mn = r + ((lane >> 4) << 3);            // i & 15
      int jcol = t * 16 + (lane & 15);            // key pos 0..127
      int ks = jcol >> 5, kk = jcol & 31;
      p_s[((wv * 4 + ks) * 32 + mn + frag_lh(kk)) * 16 + frag_j(kk)] =
          bfbits(acc[t][r] * rsum[r]);
    }
  // Each wave reads back only its own fragments; same-wave LDS ops stay
  // in order (DScnt), so no block barrier is needed here.

  // ---- O = P @ V (K = WWIN = 128, four WMMA K-steps, 64 output cols) ----
  v8f oacc[4];
  for (int t = 0; t < 4; ++t)
    for (int e = 0; e < 8; ++e) oacc[t][e] = 0.0f;

#pragma unroll
  for (int ks = 0; ks < 4; ++ks) {
    const v16bf af = ld_frag(&p_s[((wv * 4 + ks) * 32 + lane) * 16]);
#pragma unroll
    for (int t = 0; t < 4; ++t) {
      const v16bf bfr = ld_frag(&v_s[((t * 4 + ks) * 32 + lane) * 16]);
      oacc[t] = wmma_bf16(af, bfr, oacc[t]);
    }
  }

  // Un-window: token n = w*PWIN + pp ; store O[b][n][h*64+d] (bf16 bits)
  const int pp = blk & 31, h = (blk >> 5) & 7, b = blk >> 8;
  for (int t = 0; t < 4; ++t) {
    int d = t * 16 + (lane & 15);
    for (int r = 0; r < 8; ++r) {
      int wn = wv * 16 + r + ((lane >> 4) << 3);
      int n = wn * PWIN + pp;
      ow[((size_t)b * NTOK + n) * (NHEAD * DHEAD) + h * DHEAD + d] =
          bfbits(oacc[t][r]);
    }
  }
}

// ---------------------------------------------------------------------------
// Kernel 3: out = O @ Wout + bout, with roll(+DISP) folded into the store.
// grid = (4 col-tiles, 256 row-tiles), block = 256 (8 waves).
// ---------------------------------------------------------------------------
__global__ __launch_bounds__(256) void swin_out_gemm(
    const unsigned short* __restrict__ O, const unsigned short* __restrict__ wo,
    const float* __restrict__ bout, float* __restrict__ out)
{
  __shared__ __align__(32) unsigned short As[8 * 32 * 16];  // 8KB
  __shared__ __align__(32) unsigned short Bs[8 * 32 * 16];  // 8KB
  const int tid = threadIdx.x, lane = tid & 31, wv = tid >> 5;
  const int ntile = blockIdx.x;          // 0..3
  const int mtile = blockIdx.y;          // 0..255
  const int b  = (mtile * 128) / NTOK;
  const int n0 = (mtile * 128) % NTOK;

  v8f acc[8];
  for (int t = 0; t < 8; ++t)
    for (int e = 0; e < 8; ++e) acc[t][e] = 0.0f;

  for (int k0 = 0; k0 < DMODEL; k0 += 32) {
    // Stage A (bf16 O): b32 global load (2 bf16 along K) -> b32 LDS store.
    for (int i = tid; i < 128 * 16; i += 256) {
      int r = i >> 4, c = (i & 15) * 2;
      unsigned d = *(const unsigned*)&O[((size_t)b * NTOK + n0 + r) * DMODEL + k0 + c];
      int dst = (((r >> 4) * 32) + (r & 15) + frag_lh(c)) * 16 + frag_j(c);
      *(unsigned*)&As[dst] = d;
    }
    // Stage B (Wout bf16): b64 global load -> 4 b16 scattered LDS stores.
    for (int i = tid; i < 32 * 32; i += 256) {
      int c = i >> 5, jq = (i & 31) * 4;
      unsigned long long w4 =
          *(const unsigned long long*)&wo[(size_t)(k0 + c) * DMODEL + ntile * 128 + jq];
      int j = frag_j(c), lh = frag_lh(c);
#pragma unroll
      for (int e = 0; e < 4; ++e) {
        int jc = jq + e;
        Bs[(((jc >> 4) * 32) + (jc & 15) + lh) * 16 + j] =
            (unsigned short)(w4 >> (16 * e));
      }
    }
    if (k0 + 32 < DMODEL) {
      int r = tid >> 1;
      __builtin_prefetch(
          &O[((size_t)b * NTOK + n0 + r) * DMODEL + k0 + 32 + (tid & 1) * 16], 0, 0);
    }
    __syncthreads();

    const v16bf af = ld_frag(&As[(wv * 32 + lane) * 16]);
#pragma unroll
    for (int t = 0; t < 8; ++t) {
      const v16bf bfr = ld_frag(&Bs[(t * 32 + lane) * 16]);
      acc[t] = wmma_bf16(af, bfr, acc[t]);
    }
    __syncthreads();
  }

  for (int t = 0; t < 8; ++t) {
    int col = ntile * 128 + t * 16 + (lane & 15);
    float bias = bout[col];
    for (int r = 0; r < 8; ++r) {
      int m = r + ((lane >> 4) << 3);
      int n  = n0 + wv * 16 + m;           // shifted-space row
      int nf = (n + DISP) & (NTOK - 1);    // cyclic back-shift
      out[((size_t)b * NTOK + nf) * DMODEL + col] = acc[t][r] + bias;
    }
  }
}

// ---------------------------------------------------------------------------
// Host launcher
// ---------------------------------------------------------------------------
extern "C" void kernel_launch(void* const* d_in, const int* in_sizes, int n_in,
                              void* d_out, int out_size, void* d_ws, size_t ws_size,
                              hipStream_t stream) {
  (void)in_sizes; (void)n_in; (void)out_size; (void)ws_size;
  const float* x    = (const float*)d_in[0];   // [8,4096,512]
  const float* Wqkv = (const float*)d_in[1];   // [512,1536]
  const float* pos  = (const float*)d_in[2];   // [255]
  const float* Wout = (const float*)d_in[3];   // [512,512]
  const float* bout = (const float*)d_in[4];   // [512]
  float* out = (float*)d_out;

  // Workspace layout (bytes):
  //   [0,        1572864)  Wqkv bf16
  //   [1572864,  2097152)  Wout bf16
  //   then q, k, v (each 33554432 B, windowed bf16), then O (33554432 B)
  char* ws = (char*)d_ws;
  unsigned short* wqkv_b = (unsigned short*)(ws);
  unsigned short* wout_b = (unsigned short*)(ws + 1572864);
  unsigned short* qw = (unsigned short*)(ws + 2097152);
  unsigned short* kw = (unsigned short*)(ws + 2097152 + (size_t)1 * 33554432);
  unsigned short* vw = (unsigned short*)(ws + 2097152 + (size_t)2 * 33554432);
  unsigned short* ow = (unsigned short*)(ws + 2097152 + (size_t)3 * 33554432);

  swin_cvt_bf16<<<3072, 256, 0, stream>>>(Wqkv, wqkv_b, 512 * 1536);
  swin_cvt_bf16<<<1024, 256, 0, stream>>>(Wout, wout_b, 512 * 512);
  swin_qkv_gemm<<<dim3(12, 256), 256, 0, stream>>>(x, wqkv_b, qw, kw, vw);
  swin_attn<<<dim3(BDIM * NHEAD * PWIN), 256, 0, stream>>>(qw, kw, vw, pos, ow);
  swin_out_gemm<<<dim3(4, 256), 256, 0, stream>>>(ow, wout_b, bout, out);
}
